// VectorQuantize_67216238183242
// MI455X (gfx1250) — matrix operand order repaired and checked by reading, not compiled, not run on hardware
//
#include <hip/hip_runtime.h>
#include <stdint.h>

typedef __attribute__((ext_vector_type(16))) __bf16 v16bf;
typedef __attribute__((ext_vector_type(8)))  float  v8f;

#define N_ROWS 65536
#define DIM    128
#define KCODES 4096
#define DECAY  0.99f
#define OMD    0.01f   // 1 - DECAY
#define COMMIT_W 1.0f
#define EPSV   1e-5f

// ---- d_out layout (float elements), tuple order of the reference ----
#define IND_OFF  (N_ROWS*DIM)            // 8388608 : embed_ind (as float)
#define LOSS_OFF (IND_OFF + N_ROWS)      // 8454144 : commit_loss scalar
#define NE_OFF   (LOSS_OFF + 1)          // new_embed [K,D]
#define NCS_OFF  (NE_OFF + KCODES*DIM)   // new_cluster_size [K]

// ---- workspace layout (float elements) ----
#define WS_COUNTS 0                       // [K]
#define WS_ESUM   (KCODES)                // [K*D]
#define WS_LOSS   (WS_ESUM + KCODES*DIM)  // [1]
#define WS_NCS    (WS_LOSS + 1)           // [1]
#define WS_ZERON  (WS_NCS + 1)            // floats to zero each call
#define WS_EHI    (WS_ZERON + 6)          // packed bf16-hi pairs, K*D/2 u32
#define WS_ELO    (WS_EHI + KCODES*DIM/2) // packed bf16-lo pairs
#define WS_ESQ    (WS_ELO + KCODES*DIM/2) // [K] |e|^2

union FragBF { v16bf v; uint32_t u[8]; uint4 q[2]; };

__device__ __forceinline__ uint16_t f2bf(float f) {
  uint32_t u = __float_as_uint(f);
  // round-to-nearest-even bf16
  uint32_t r = (u + 0x7FFFu + ((u >> 16) & 1u)) >> 16;
  return (uint16_t)r;
}
__device__ __forceinline__ float bf2f(uint16_t h) {
  return __uint_as_float(((uint32_t)h) << 16);
}
// split two consecutive fp32 into packed bf16 hi-pair and lo-pair words
__device__ __forceinline__ void split2(float a, float b, uint32_t& hi, uint32_t& lo) {
  uint16_t ha = f2bf(a), hb = f2bf(b);
  uint16_t la = f2bf(a - bf2f(ha)), lb = f2bf(b - bf2f(hb));
  hi = (uint32_t)ha | ((uint32_t)hb << 16);
  lo = (uint32_t)la | ((uint32_t)lb << 16);
}

__device__ __forceinline__ v8f wmma_bf16(const FragBF& a, const FragBF& b, v8f c) {
  return __builtin_amdgcn_wmma_f32_16x16x32_bf16(false, a.v, false, b.v,
                                                 (short)0, c, false, false);
}

// --------------------------------------------------------------------
// Kernel 0: zero the accumulators in ws (counts, embed_sum, loss, ncs_sum)
// --------------------------------------------------------------------
__global__ void vq_zero(float* __restrict__ ws) {
  int i = blockIdx.x * 256 + threadIdx.x;
  if (i < WS_ZERON) ws[i] = 0.0f;
}

// --------------------------------------------------------------------
// Kernel 1: pack embed -> bf16 hi/lo pair-words + e_sq.  One block per code.
// --------------------------------------------------------------------
__global__ __launch_bounds__(64)
void vq_prep(const float* __restrict__ embed,
             uint32_t* __restrict__ ehi, uint32_t* __restrict__ elo,
             float* __restrict__ esq) {
  __shared__ float sred[2];
  const int k = blockIdx.x;
  const int t = threadIdx.x;              // 0..63, one bf16 pair each
  float2 v = *(const float2*)(embed + (size_t)k * DIM + t * 2);
  uint32_t hi, lo;
  split2(v.x, v.y, hi, lo);
  ehi[(size_t)k * 64 + t] = hi;
  elo[(size_t)k * 64 + t] = lo;
  float s = v.x * v.x + v.y * v.y;
#pragma unroll
  for (int off = 16; off; off >>= 1) s += __shfl_xor(s, off, 32);
  if ((t & 31) == 0) sred[t >> 5] = s;
  __syncthreads();
  if (t == 0) esq[k] = sred[0] + sred[1];
}

// --------------------------------------------------------------------
// Kernel 2: 256 rows per block (8 waves x 2 row-tiles), all K codes.
// Each wave holds TWO 16x32-chunk A fragment sets in registers so every
// B fragment fetched from LDS feeds 24 WMMAs (2x compute per LDS byte),
// and uses 4 independent accumulator chains for WMMA ILP.
// --------------------------------------------------------------------
__global__ __launch_bounds__(256)
void vq_assign(const float* __restrict__ x,
               const float* __restrict__ embed,
               const uint32_t* __restrict__ ehi,
               const uint32_t* __restrict__ elo,
               const float* __restrict__ esq,
               float* __restrict__ out,
               float* __restrict__ counts,
               float* __restrict__ esum,
               float* __restrict__ loss_acc) {
  __shared__ uint32_t sBh[16 * 64];   // 16 codes x 64 hi pair-words (4 KB)
  __shared__ uint32_t sBl[16 * 64];   // lo pair-words               (4 KB)
  __shared__ int sIdx[256];

  const int tid  = threadIdx.x;
  const int wave = tid >> 5;
  const int lane = tid & 31;
  const int half = lane >> 4;          // A/B lane-half (K sub-group)
  const int m    = lane & 15;          // A row / B column within tile
  const int rowbase = blockIdx.x * 256;

  // ---- build A fragments (16x32 bf16 layout, ISA 7.12.2) ----
  // 2 row-tiles x 4 K-chunks; lane-half 0 holds K {0..7,16..23}, half 1 the rest
  FragBF ahi[2][4], alo[2][4];
#pragma unroll
  for (int T = 0; T < 2; ++T) {
    const float* xr = x + ((size_t)rowbase + wave * 32 + T * 16 + m) * DIM;
#pragma unroll
    for (int c = 0; c < 4; ++c) {
#pragma unroll
      for (int g = 0; g < 2; ++g) {      // g=0 -> VGPR0-3, g=1 -> VGPR4-7
        const float* p = xr + c * 32 + g * 16 + half * 8;
        float4 f0 = *(const float4*)(p);
        float4 f1 = *(const float4*)(p + 4);
        float f[8] = {f0.x, f0.y, f0.z, f0.w, f1.x, f1.y, f1.z, f1.w};
#pragma unroll
        for (int pr = 0; pr < 4; ++pr)
          split2(f[2 * pr], f[2 * pr + 1],
                 ahi[T][c].u[g * 4 + pr], alo[T][c].u[g * 4 + pr]);
      }
    }
  }

  float bestv[2][8];
  int   besti[2][8];
#pragma unroll
  for (int T = 0; T < 2; ++T)
#pragma unroll
    for (int r = 0; r < 8; ++r) { bestv[T][r] = 3.4e38f; besti[T][r] = 0; }

  // software-pipelined staging of embed k-tiles (16 codes) through LDS
  uint4 ph = *(const uint4*)(ehi + tid * 4);
  uint4 pl = *(const uint4*)(elo + tid * 4);

  for (int t = 0; t < KCODES / 16; ++t) {
    __syncthreads();
    *(uint4*)&sBh[tid * 4] = ph;
    *(uint4*)&sBl[tid * 4] = pl;
    __syncthreads();
    const int t2 = (t + 1 < KCODES / 16) ? (t + 1) : t;
    ph = *(const uint4*)(ehi + t2 * 1024 + tid * 4);
    pl = *(const uint4*)(elo + t2 * 1024 + tid * 4);

    const int kb = t * 16;
    const float en = esq[kb + m];
    v8f acc[2][2];
#pragma unroll
    for (int T = 0; T < 2; ++T)
#pragma unroll
      for (int h = 0; h < 2; ++h)
        acc[T][h] = (v8f){0.f, 0.f, 0.f, 0.f, 0.f, 0.f, 0.f, 0.f};

#pragma unroll
    for (int c = 0; c < 4; ++c) {
      // B 32x16 bf16: lane-half h holds K {8h..8h+7, 16+8h..16+8h+7} of the chunk
      FragBF bh, bl;
      const int base = m * 64 + c * 16 + half * 4;
      bh.q[0] = *(const uint4*)&sBh[base];
      bh.q[1] = *(const uint4*)&sBh[base + 8];
      bl.q[0] = *(const uint4*)&sBl[base];
      bl.q[1] = *(const uint4*)&sBl[base + 8];
      const int h = c >> 1;   // chunk-split accumulator: 4 independent chains
      // hi*hi + hi*lo + lo*hi  ~= fp32-accurate x.e, fp32 accumulate
      acc[0][h] = wmma_bf16(ahi[0][c], bh, acc[0][h]);
      acc[1][h] = wmma_bf16(ahi[1][c], bh, acc[1][h]);
      acc[0][h] = wmma_bf16(ahi[0][c], bl, acc[0][h]);
      acc[1][h] = wmma_bf16(ahi[1][c], bl, acc[1][h]);
      acc[0][h] = wmma_bf16(alo[0][c], bh, acc[0][h]);
      acc[1][h] = wmma_bf16(alo[1][c], bh, acc[1][h]);
    }
    // score = |e|^2 - 2 x.e   (|x|^2 is constant per row -> irrelevant to argmin)
#pragma unroll
    for (int T = 0; T < 2; ++T)
#pragma unroll
      for (int r = 0; r < 8; ++r) {
        float s = en - 2.0f * (acc[T][0][r] + acc[T][1][r]);
        if (s < bestv[T][r]) { bestv[T][r] = s; besti[T][r] = kb + m; }
      }
  }

  // cross-lane argmin within each 16-lane half (rows r for half0, 8+r for half1)
#pragma unroll
  for (int T = 0; T < 2; ++T)
#pragma unroll
    for (int r = 0; r < 8; ++r) {
      float v = bestv[T][r];
      int   i = besti[T][r];
#pragma unroll
      for (int off = 1; off < 16; off <<= 1) {
        float v2 = __shfl_xor(v, off, 32);
        int   i2 = __shfl_xor(i, off, 32);
        if (v2 < v || (v2 == v && i2 < i)) { v = v2; i = i2; }
      }
      if (m == 0) {
        const int mrow = wave * 32 + T * 16 + half * 8 + r;
        sIdx[mrow] = i;
        out[IND_OFF + rowbase + mrow] = (float)i;
      }
    }
  __syncthreads();

  // ---- gather quantize, commit loss partial, scatter-add EMA stats ----
  const int rl   = tid;            // one thread per row (256 rows)
  const int code = sIdx[rl];
  const size_t grow = (size_t)rowbase + rl;
  const float* ev = embed + (size_t)code * DIM;
  const float* xv = x + grow * DIM;
  float* qv = out + grow * DIM;
  float* es = esum + (size_t)code * DIM;
  float csum = 0.0f;
#pragma unroll 4
  for (int j = 0; j < 32; ++j) {
    float4 e4 = *(const float4*)(ev + j * 4);
    float4 x4 = *(const float4*)(xv + j * 4);
    *(float4*)(qv + j * 4) = e4;            // quantize_st == quantize numerically
    float dx = e4.x - x4.x, dy = e4.y - x4.y, dz = e4.z - x4.z, dw = e4.w - x4.w;
    csum += dx * dx + dy * dy + dz * dz + dw * dw;
    atomicAdd(es + j * 4 + 0, x4.x);
    atomicAdd(es + j * 4 + 1, x4.y);
    atomicAdd(es + j * 4 + 2, x4.z);
    atomicAdd(es + j * 4 + 3, x4.w);
  }
  atomicAdd(&counts[code], 1.0f);
#pragma unroll
  for (int off = 16; off; off >>= 1) csum += __shfl_xor(csum, off, 32);
  if (lane == 0) atomicAdd(loss_acc, csum);
}

// --------------------------------------------------------------------
// Kernel 3: new_cluster_size + its total (for Laplace smoothing)
// --------------------------------------------------------------------
__global__ __launch_bounds__(256)
void vq_cluster(const float* __restrict__ cs, const float* __restrict__ counts,
                float* __restrict__ out, float* __restrict__ ncs_sum) {
  __shared__ float sred[8];
  const int k = blockIdx.x * 256 + threadIdx.x;
  float v = cs[k] * DECAY + counts[k] * OMD;
  out[NCS_OFF + k] = v;
  float s = v;
#pragma unroll
  for (int off = 16; off; off >>= 1) s += __shfl_xor(s, off, 32);
  if ((threadIdx.x & 31) == 0) sred[threadIdx.x >> 5] = s;
  __syncthreads();
  if (threadIdx.x == 0) {
    float tot = 0.f;
#pragma unroll
    for (int w = 0; w < 8; ++w) tot += sred[w];
    atomicAdd(ncs_sum, tot);
  }
}

// --------------------------------------------------------------------
// Kernel 4: commit loss scalar + new_embed = EMA(embed_avg)/smoothed
// --------------------------------------------------------------------
__global__ __launch_bounds__(256)
void vq_finalize(const float* __restrict__ embed_avg, const float* __restrict__ esum,
                 const float* __restrict__ loss_acc, const float* __restrict__ ncs_sum,
                 float* __restrict__ out) {
  const int idx = blockIdx.x * 256 + threadIdx.x;   // over K*D
  if (idx == 0)
    out[LOSS_OFF] = loss_acc[0] * (COMMIT_W / (float)((size_t)N_ROWS * DIM));
  const int k = idx / DIM;
  const float ntot = ncs_sum[0];
  const float ncs  = out[NCS_OFF + k];
  const float smoothed = (ncs + EPSV) / (ntot + (float)KCODES * EPSV) * ntot;
  const float ea = embed_avg[idx] * DECAY + esum[idx] * OMD;
  out[NE_OFF + idx] = ea / smoothed;
}

extern "C" void kernel_launch(void* const* d_in, const int* in_sizes, int n_in,
                              void* d_out, int out_size, void* d_ws, size_t ws_size,
                              hipStream_t stream) {
  const float* x     = (const float*)d_in[0];
  const float* embed = (const float*)d_in[1];
  const float* cs    = (const float*)d_in[2];
  const float* eavg  = (const float*)d_in[3];
  float* out = (float*)d_out;
  float* ws  = (float*)d_ws;

  float*    counts = ws + WS_COUNTS;
  float*    esum   = ws + WS_ESUM;
  float*    loss   = ws + WS_LOSS;
  float*    ncssum = ws + WS_NCS;
  uint32_t* ehi    = (uint32_t*)(ws + WS_EHI);
  uint32_t* elo    = (uint32_t*)(ws + WS_ELO);
  float*    esq    = ws + WS_ESQ;

  vq_zero<<<(WS_ZERON + 255) / 256, 256, 0, stream>>>(ws);
  vq_prep<<<KCODES, 64, 0, stream>>>(embed, ehi, elo, esq);
  vq_assign<<<N_ROWS / 256, 256, 0, stream>>>(x, embed, ehi, elo, esq,
                                              out, counts, esum, loss);
  vq_cluster<<<KCODES / 256, 256, 0, stream>>>(cs, counts, out, ncssum);
  vq_finalize<<<(KCODES * DIM) / 256, 256, 0, stream>>>(eavg, esum, loss, ncssum, out);
}
